// VariableFilter_50190987821778
// MI455X (gfx1250) — matrix operand order repaired and checked by reading, not compiled, not run on hardware
//
#include <hip/hip_runtime.h>
#include <hip/hip_bf16.h>

typedef __attribute__((ext_vector_type(16))) _Float16 v16h;
typedef __attribute__((ext_vector_type(8)))  _Float16 v8h;
typedef __attribute__((ext_vector_type(8)))  float    v8f;

#define BB 32
#define CC 1024
#define DD 128
#define RR (BB*CC)      // 32768 rows
#define TOPK 32

static __device__ __forceinline__ v16h wmma_load_a(const _Float16* rowptr, int koff) {
    // A 16x32 f16 fragment, lane l holds row m=l&15:
    //   lanes 0-15 : K = koff+{0..15}
    //   lanes16-31 : K = koff+{0..15}   (koff already includes hi*16)
    v8h lo = *(const v8h*)(rowptr + koff);
    v8h hi = *(const v8h*)(rowptr + koff + 8);
    return __builtin_shufflevector(lo, hi, 0,1,2,3,4,5,6,7,8,9,10,11,12,13,14,15);
}

static __device__ __forceinline__ v8f wmma_f16(v16h a, v16h b, v8f c) {
    return __builtin_amdgcn_wmma_f32_16x16x32_f16(false, a, false, b, (short)0, c, false, false);
}

// ---------------- K0: M = Wq^T Wk (f16), u = Wq^T bk, v = Wk^T bq, cconst = bq.bk
__global__ __launch_bounds__(256) void k0_prep(const float* __restrict__ Wq,
                                               const float* __restrict__ Wk,
                                               const float* __restrict__ bq,
                                               const float* __restrict__ bk,
                                               _Float16* __restrict__ Mf16,
                                               float* __restrict__ u,
                                               float* __restrict__ v,
                                               float* __restrict__ cconst) {
    int idx = blockIdx.x * 256 + threadIdx.x;       // 64 blocks -> 16384 = 128*128
    int i = idx >> 7, m = idx & 127;
    float s = 0.f;
    for (int j = 0; j < DD; ++j) s += Wq[j*DD + i] * Wk[j*DD + m];
    Mf16[idx] = (_Float16)s;
    if (blockIdx.x == 0) {
        int t = threadIdx.x;
        if (t < 128) {
            float su = 0.f;
            for (int j = 0; j < DD; ++j) su += Wq[j*DD + t] * bk[j];
            u[t] = su;
        } else {
            int mm = t - 128;
            float sv = 0.f;
            for (int j = 0; j < DD; ++j) sv += Wk[j*DD + mm] * bq[j];
            v[mm] = sv;
        }
        if (t == 0) {
            float sc = 0.f;
            for (int j = 0; j < DD; ++j) sc += bq[j] * bk[j];
            cconst[0] = sc;
        }
    }
}

// ---------------- K1a: hh = f16(h); sq = sum h^2; uc = h.u; ve = h.v  (one wave per row)
__global__ __launch_bounds__(256) void k1a_rows(const float* __restrict__ h,
                                                const float* __restrict__ u,
                                                const float* __restrict__ v,
                                                _Float16* __restrict__ hh,
                                                float* __restrict__ sq,
                                                float* __restrict__ uc,
                                                float* __restrict__ ve) {
    int wave = threadIdx.x >> 5, lane = threadIdx.x & 31;
    int row  = blockIdx.x * 8 + wave;
    if (row >= RR) return;
    const float* rp = h + (size_t)row * DD;
    _Float16*    op = hh + (size_t)row * DD;
    float s2 = 0.f, su = 0.f, sv = 0.f;
    #pragma unroll
    for (int i = 0; i < 4; ++i) {
        int j = lane + (i << 5);
        float x = rp[j];
        op[j] = (_Float16)x;
        s2 += x * x;
        su += x * u[j];
        sv += x * v[j];
    }
    #pragma unroll
    for (int off = 16; off > 0; off >>= 1) {
        s2 += __shfl_xor(s2, off, 32);
        su += __shfl_xor(su, off, 32);
        sv += __shfl_xor(sv, off, 32);
    }
    if (lane == 0) { sq[row] = s2; uc[row] = su; ve[row] = sv; }
}

// ---------------- K1b: G = hh @ M^T (f16 out) via WMMA. B(k,n)=M[n][k] -> contiguous row of Mf16.
__global__ __launch_bounds__(256) void k1b_gemm(const _Float16* __restrict__ hh,
                                                const _Float16* __restrict__ Mf16,
                                                _Float16* __restrict__ G) {
    int wave = threadIdx.x >> 5, lane = threadIdx.x & 31;
    int row0 = blockIdx.x * 16;                 // 2048 blocks cover 32768 rows
    int m    = lane & 15, hi = lane >> 4;
    int ncol = wave * 16 + m;                   // 8 waves x 16 = 128 output cols
    const _Float16* arow = hh + (size_t)(row0 + m) * DD;
    const _Float16* brow = Mf16 + (size_t)ncol * DD;
    v8f acc = {};
    #pragma unroll
    for (int kk = 0; kk < 4; ++kk) {
        v16h a = wmma_load_a(arow, kk * 32 + hi * 16);
        v16h b = *(const v16h*)(brow + kk * 32 + hi * 16);
        acc = wmma_f16(a, b, acc);
    }
    #pragma unroll
    for (int r = 0; r < 8; ++r) {
        int mm = r + hi * 8;
        G[(size_t)(row0 + mm) * DD + ncol] = (_Float16)acc[r];
    }
}

// ---------------- K2: fused scores. Per workgroup: batch b, 16-row c-tile; each wave: 16x128 e-strip.
__global__ __launch_bounds__(256) void k2_scores(const _Float16* __restrict__ hh,
                                                 const _Float16* __restrict__ G,
                                                 const float* __restrict__ sq,
                                                 const float* __restrict__ uc,
                                                 const float* __restrict__ ve,
                                                 const float* __restrict__ cconst,
                                                 float* __restrict__ scores) {
    const float INV_SQRT_D = 0.08838834764831845f;   // 1/sqrt(128)
    const float INV_2SIG   = 1.0f / (2.0f + 1e-8f);  // 1/(2*sigma^2+1e-8)
    int wave = threadIdx.x >> 5, lane = threadIdx.x & 31;
    int b  = blockIdx.y;
    int c0 = blockIdx.x * 16;
    int n  = lane & 15, hi = lane >> 4;
    const _Float16* hb = hh + (size_t)b * CC * DD;
    const _Float16* Gb = G  + (size_t)b * CC * DD;
    const float*    sqb = sq + (size_t)b * CC;
    const float*    ucb = uc + (size_t)b * CC;
    const float*    veb = ve + (size_t)b * CC;
    float cc = cconst[0];

    // A fragments (shared across both GEMMs and all 8 e-subtiles)
    const _Float16* arow = hb + (size_t)(c0 + n) * DD;
    v16h afrag[4];
    #pragma unroll
    for (int kk = 0; kk < 4; ++kk) afrag[kk] = wmma_load_a(arow, kk * 32 + hi * 16);

    float sqc[8], ucc[8];
    #pragma unroll
    for (int r = 0; r < 8; ++r) {
        int mrow = c0 + r + hi * 8;
        sqc[r] = sqb[mrow];
        ucc[r] = ucb[mrow];
    }

    #pragma unroll
    for (int t = 0; t < 8; ++t) {
        int e0 = wave * 128 + t * 16;
        int eg = e0 + n;                        // this lane's global column
        float sqe = sqb[eg];
        float vee = veb[eg];
        const _Float16* grow = Gb + (size_t)eg * DD;
        const _Float16* hrow = hb + (size_t)eg * DD;
        v8f acc1 = {};                          // h_c . (M h_e)  (bilinear)
        v8f acc2 = {};                          // h_c . h_e      (inner)
        #pragma unroll
        for (int kk = 0; kk < 4; ++kk) {
            v16h bg = *(const v16h*)(grow + kk * 32 + hi * 16);
            v16h bh = *(const v16h*)(hrow + kk * 32 + hi * 16);
            acc1 = wmma_f16(afrag[kk], bg, acc1);
            acc2 = wmma_f16(afrag[kk], bh, acc2);
        }
        #pragma unroll
        for (int r = 0; r < 8; ++r) {
            int cg = c0 + r + hi * 8;
            float lear = (acc1[r] + ucc[r] + vee + cc) * INV_SQRT_D;
            float d2 = fmaxf(sqc[r] + sqe - 2.0f * acc2[r], 0.0f);
            float sim = __logf(__expf(-d2 * INV_2SIG) + 1e-8f);
            float sc = 0.3f * lear + 0.7f * sim;
            if (cg == eg) sc = -__builtin_huge_valf();
            scores[((size_t)b * CC + cg) * CC + eg] = sc;
        }
    }
}

// ---------------- K3: top-32 per row, one wave32 per row, lex-max selection (stable ties)
__global__ __launch_bounds__(256) void k3_topk(const float* __restrict__ scores,
                                               int* __restrict__ ti,
                                               float* __restrict__ ts) {
    int wave = threadIdx.x >> 5, lane = threadIdx.x & 31;
    int row  = blockIdx.x * 8 + wave;
    if (row >= RR) return;
    const float* rp = scores + (size_t)row * CC;
    float vals[32];
    #pragma unroll
    for (int j = 0; j < 32; ++j) vals[j] = rp[lane + (j << 5)];

    for (int p = 0; p < TOPK; ++p) {
        float bv = -__builtin_huge_valf();
        int bidx = 0x7fffffff;
        #pragma unroll
        for (int j = 0; j < 32; ++j) {
            int idx = lane + (j << 5);
            float vv = vals[j];
            if (vv > bv || (vv == bv && idx < bidx)) { bv = vv; bidx = idx; }
        }
        #pragma unroll
        for (int off = 16; off > 0; off >>= 1) {
            float ov = __shfl_xor(bv, off, 32);
            int   oi = __shfl_xor(bidx, off, 32);
            if (ov > bv || (ov == bv && oi < bidx)) { bv = ov; bidx = oi; }
        }
        if (lane == 0) {
            ti[(size_t)row * TOPK + p] = bidx;
            ts[(size_t)row * TOPK + p] = bv;
        }
        if ((bidx & 31) == lane) {
            int jj = bidx >> 5;
            #pragma unroll
            for (int j = 0; j < 32; ++j)
                if (j == jj) vals[j] = -__builtin_huge_valf();
        }
    }
}

extern "C" void kernel_launch(void* const* d_in, const int* in_sizes, int n_in,
                              void* d_out, int out_size, void* d_ws, size_t ws_size,
                              hipStream_t stream) {
    const float* h  = (const float*)d_in[0];
    const float* Wq = (const float*)d_in[1];
    const float* bq = (const float*)d_in[2];
    const float* Wk = (const float*)d_in[3];
    const float* bk = (const float*)d_in[4];

    // workspace carve (256B aligned)
    char* ws = (char*)d_ws;
    size_t o = 0;
    auto carve = [&](size_t bytes) { char* p = ws + o; o = (o + bytes + 255) & ~(size_t)255; return p; };
    _Float16* Mf16 = (_Float16*)carve((size_t)DD * DD * 2);
    float*    u    = (float*)carve(DD * 4);
    float*    v    = (float*)carve(DD * 4);
    float*    cc   = (float*)carve(256);
    _Float16* hh   = (_Float16*)carve((size_t)RR * DD * 2);
    _Float16* G    = (_Float16*)carve((size_t)RR * DD * 2);
    float*    sq   = (float*)carve((size_t)RR * 4);
    float*    uc   = (float*)carve((size_t)RR * 4);
    float*    ve   = (float*)carve((size_t)RR * 4);
    (void)ws_size;

    // output carve: [topk_idx int32][topk_scores f32][scores f32]
    const size_t nTop = (size_t)BB * CC * TOPK;     // 1,048,576
    int*   ti     = (int*)d_out;
    float* ts     = (float*)d_out + nTop;
    float* scores = (float*)d_out + 2 * nTop;

    k0_prep  <<<dim3(64),        dim3(256), 0, stream>>>(Wq, Wk, bq, bk, Mf16, u, v, cc);
    k1a_rows <<<dim3(RR / 8),    dim3(256), 0, stream>>>(h, u, v, hh, sq, uc, ve);
    k1b_gemm <<<dim3(RR / 16),   dim3(256), 0, stream>>>(hh, Mf16, G);
    k2_scores<<<dim3(CC / 16, BB), dim3(256), 0, stream>>>(hh, G, sq, uc, ve, cc, scores);
    k3_topk  <<<dim3(RR / 8),    dim3(256), 0, stream>>>(scores, ti, ts);
}
